// RelMemRNN_77481210020578
// MI455X (gfx1250) — matrix operand order repaired and checked by reading, not compiled, not run on hardware
//
#include <hip/hip_runtime.h>
#include <math.h>

typedef _Float16 half_t;
typedef __attribute__((ext_vector_type(16))) _Float16 v16h;
typedef __attribute__((ext_vector_type(8)))  _Float16 v8h;
typedef __attribute__((ext_vector_type(8)))  float    v8f;

#define T_LEN  256
#define B_SZ   128
#define INP_D  512
#define HID_D  512
#define KSHORT 32
#define RLONG  64
#define NSLOT  (KSHORT + RLONG)
#define NEGM   (-1e30f)

// ---------------------------------------------------------------------------
// WMMA GEMM, LDS-free: C[M=128, N] = act(A[M,K] @ W[K,N] + bias)
//   A: row-major f16, split across two sources (A0 rows for k<K0, A1 after) --
//      per-lane fragment = two contiguous 8-half chunks -> global b128 loads.
//   W: stored TRANSPOSED (WT[n*K + k], f16, built once in precompute) so the
//      B fragment (contiguous along K) is also two global b128 loads.
// Block = 128 threads (4 waves); block tile 16(M) x 128(N); each wave owns
// two 16x16 accumulators sharing one A fragment.  K/K0 are template params,
// so the A0/A1 segments are separate statically-bounded loops (no per-iter
// pointer select in the pipelined body).
// ---------------------------------------------------------------------------
struct GemmP {
  const half_t* A0; const half_t* A1;
  const half_t* WT; int N;
  const float* bias;
  float* o32r;   // raw f32 out (va, PH)
  float* o32t;   // tanh f32 out (d_out rows)
  half_t* o16t;  // tanh f16 out (h16)
  half_t* o16r;  // raw f16 out (st16 at t=0)
};

__device__ __forceinline__ v16h load_frag_a(const half_t* p) {
  // ISA 16-bit A 16x32 layout: lane holds K = kb..kb+7 and kb+16..kb+23
  const v8h lo = *(const v8h*)p;
  const v8h hi = *(const v8h*)(p + 16);
  return __builtin_shufflevector(lo, hi, 0,1,2,3,4,5,6,7,8,9,10,11,12,13,14,15);
}
__device__ __forceinline__ v16h load_frag_b(const half_t* p) {
  // ISA 16-bit B 32x16 layout: lane holds K = kb..kb+7 and kb+16..kb+23
  const v8h lo = *(const v8h*)p;
  const v8h hi = *(const v8h*)(p + 16);
  return __builtin_shufflevector(lo, hi, 0,1,2,3,4,5,6,7,8,9,10,11,12,13,14,15);
}

template<int K, int K0>
__device__ __forceinline__ void gemm_tile(const GemmP p, int bx, int by) {
  const int lane = threadIdx.x & 31;
  const int wv   = threadIdx.x >> 5;          // 0..3
  const int bm   = by * 16;
  const int bn   = bx * 128;
  const int row  = lane & 15;
  const int kbA  = (lane >> 4) * 8;
  const int kbB  = (lane >> 4) * 8;
  const int col0 = bn + wv * 32 + (lane & 15);
  const int col1 = col0 + 16;

  const half_t* a0p = p.A0 + (size_t)(bm + row) * K0 + kbA;
  const half_t* b0p = p.WT + (size_t)col0 * K + kbB;
  const half_t* b1p = p.WT + (size_t)col1 * K + kbB;

  v8f acc0 = {0.f,0.f,0.f,0.f,0.f,0.f,0.f,0.f};
  v8f acc1 = {0.f,0.f,0.f,0.f,0.f,0.f,0.f,0.f};

  // ---- A0 segment: k in [0, K0) ----
#pragma unroll 4
  for (int k0 = 0; k0 < K0; k0 += 32) {
    const v16h a  = load_frag_a(a0p + k0);
    const v16h b0 = load_frag_b(b0p + k0);
    const v16h b1 = load_frag_b(b1p + k0);
    acc0 = __builtin_amdgcn_wmma_f32_16x16x32_f16(false, a, false, b0,
                                                  (short)0, acc0, false, false);
    acc1 = __builtin_amdgcn_wmma_f32_16x16x32_f16(false, a, false, b1,
                                                  (short)0, acc1, false, false);
  }
  // ---- A1 segment: k in [K0, K) ----
  if (K > K0) {
    const half_t* a1p = p.A1 + (size_t)(bm + row) * (K - K0) + kbA;
#pragma unroll 4
    for (int k0 = K0; k0 < K; k0 += 32) {
      const v16h a  = load_frag_a(a1p + (k0 - K0));
      const v16h b0 = load_frag_b(b0p + k0);
      const v16h b1 = load_frag_b(b1p + k0);
      acc0 = __builtin_amdgcn_wmma_f32_16x16x32_f16(false, a, false, b0,
                                                    (short)0, acc0, false, false);
      acc1 = __builtin_amdgcn_wmma_f32_16x16x32_f16(false, a, false, b1,
                                                    (short)0, acc1, false, false);
    }
  }

  // ---- epilogue: C layout VGPR r -> M = r (lanes<16) / r+8 (lanes>=16) ----
  const int rbase = bm + ((lane >> 4) * 8);
#pragma unroll
  for (int r = 0; r < 8; ++r) {
    const int rrow = rbase + r;
#pragma unroll
    for (int c = 0; c < 2; ++c) {
      const int ccol = c ? col1 : col0;
      float vv = c ? acc1[r] : acc0[r];
      if (p.bias) vv += p.bias[ccol];
      const float tv = tanhf(vv);
      const size_t off = (size_t)rrow * p.N + ccol;
      if (p.o32r) p.o32r[off] = vv;
      if (p.o32t) p.o32t[off] = tv;
      if (p.o16t) p.o16t[off] = (half_t)tv;
      if (p.o16r) p.o16r[off] = (half_t)vv;
    }
  }
}

template<int K, int K0>
__global__ __launch_bounds__(128) void gemm_kernel(GemmP p) {
  const int nbx = p.N >> 7;
  gemm_tile<K, K0>(p, blockIdx.x % nbx, blockIdx.x / nbx);
}

// Two independent GEMMs per launch: trims graph-replay latency on the chain.
template<int K1, int K01, int K2, int K02>
__global__ __launch_bounds__(128) void dual_gemm_kernel(GemmP p1, int nb1, GemmP p2) {
  if ((int)blockIdx.x < nb1) {
    const int nbx = p1.N >> 7;
    gemm_tile<K1, K01>(p1, blockIdx.x % nbx, blockIdx.x / nbx);
  } else {
    const int bI = blockIdx.x - nb1;
    const int nbx = p2.N >> 7;
    gemm_tile<K2, K02>(p2, bI % nbx, bI / nbx);
  }
}

// ---------------------------------------------------------------------------
// Analytic circular-buffer time: slot s holds h_tau, tau = newest time < t
// with tau % 32 == s; valid iff tau >= 0.
// ---------------------------------------------------------------------------
__device__ __forceinline__ int slot_time(int s, int t) {
  const int d = (t - 1 - s) & 31;
  return (t - 1) - d;
}

// ---------------------------------------------------------------------------
// Attention + st_new + long_scores accumulation + eviction.
// One block per batch column b -> owns all [., b] state, no atomics.
// ---------------------------------------------------------------------------
__global__ __launch_bounds__(256) void attn_kernel(
    const float* __restrict__ va, const float* __restrict__ vvec,
    const float* __restrict__ hs,          // d_out [T,B,H]
    const float* __restrict__ PH,          // [32,B,H] cached h@Uaw ring
    float* __restrict__ PL,                // [64,B,H] cached long proj
    float* __restrict__ long_mem,          // [64,B,H]
    float* __restrict__ long_mask,         // [64,B]
    float* __restrict__ long_scores,       // [T,B]
    float* __restrict__ buck,              // [B,64]
    int* __restrict__ fcnt,                // [B]
    half_t* __restrict__ st16,             // [B,H] out: st_new (f16)
    int t)
{
  const int b    = blockIdx.x;
  const int tid  = threadIdx.x;
  const int lane = tid & 31;
  const int wv   = tid >> 5;
  const size_t BH = (size_t)B_SZ * HID_D;

  __shared__ float vaS[HID_D];
  __shared__ float alphaS[NSLOT];
  __shared__ int doS, posS;

  for (int i = tid; i < HID_D; i += 256) vaS[i] = va[(size_t)b * HID_D + i];
  __syncthreads();

  // ---- scores: es[m] = v . tanh(va + P[m]) (+mask), NEGM for dead slots ----
  for (int m = wv; m < NSLOT; m += 8) {
    const float* P = nullptr;
    float msk = 0.f;
    bool dead = false;
    if (m < KSHORT) {
      if (slot_time(m, t) < 0) dead = true;
      else P = PH + (size_t)m * BH + (size_t)b * HID_D;
    } else {
      const int r = m - KSHORT;
      msk = long_mask[(size_t)r * B_SZ + b];
      if (msk < -1e29f) dead = true;            // empty bucket: skip proj read
      else P = PL + (size_t)r * BH + (size_t)b * HID_D;
    }
    if (dead) { if (lane == 0) alphaS[m] = NEGM; continue; }
    float part = 0.f;
    for (int i = lane; i < HID_D; i += 32)
      part += vvec[i] * tanhf(vaS[i] + P[i]);
#pragma unroll
    for (int off = 16; off; off >>= 1) part += __shfl_xor(part, off, 32);
    if (lane == 0) alphaS[m] = part + msk;
  }
  __syncthreads();

  // ---- softmax over 96 (wave 0, 3 values/lane, xor-reduce) ----
  if (wv == 0) {
    const float e0 = alphaS[lane], e1 = alphaS[lane + 32], e2 = alphaS[lane + 64];
    float mx = fmaxf(e0, fmaxf(e1, e2));
#pragma unroll
    for (int off = 16; off; off >>= 1) mx = fmaxf(mx, __shfl_xor(mx, off, 32));
    const float x0 = expf(e0 - mx), x1 = expf(e1 - mx), x2 = expf(e2 - mx);
    float sm = x0 + x1 + x2;
#pragma unroll
    for (int off = 16; off; off >>= 1) sm += __shfl_xor(sm, off, 32);
    const float inv = 1.f / sm;
    alphaS[lane] = x0 * inv; alphaS[lane + 32] = x1 * inv; alphaS[lane + 64] = x2 * inv;
  }
  __syncthreads();

  // ---- ct = sum_m alpha[m] * comb[m];  st_new = 0.5*(h_{t-1} + ct) ----
  for (int i = tid; i < HID_D; i += 256) {
    float acc = 0.f;
    for (int m = 0; m < NSLOT; ++m) {
      const float a = alphaS[m];
      if (a > 0.f) {
        const float* rowp = (m < KSHORT)
            ? hs + (size_t)slot_time(m, t) * BH + (size_t)b * HID_D
            : long_mem + (size_t)(m - KSHORT) * BH + (size_t)b * HID_D;
        acc += a * rowp[i];
      }
    }
    const float stn =
        0.5f * (hs[(size_t)(t - 1) * BH + (size_t)b * HID_D + i] + acc);
    st16[(size_t)b * HID_D + i] = (half_t)stn;
  }

  // ---- long_scores[tau(s), b] += alpha[s] for valid short slots ----
  if (wv == 0) {
    const int tau = slot_time(lane, t);
    if (tau >= 0) long_scores[(size_t)tau * B_SZ + b] += alphaS[lane];
  }
  __syncthreads();

  // ---- conditional bucket eviction of h_{t-32} ----
  if (t >= KSHORT) {
    if (tid == 0) {
      const float score_t = long_scores[(size_t)(t - KSHORT) * B_SZ + b];
      const int fc = fcnt[b];
      int minp = 0; float mn = buck[(size_t)b * RLONG];
      for (int q = 1; q < RLONG; ++q) {
        const float qq = buck[(size_t)b * RLONG + q];
        if (qq < mn) { mn = qq; minp = q; }
      }
      const bool not_full = fc < RLONG;
      const bool go = not_full || (score_t > mn);
      const int pos = not_full ? fc : minp;
      doS = go ? 1 : 0; posS = pos;
      if (go) {
        long_mask[(size_t)pos * B_SZ + b] = 0.f;
        buck[(size_t)b * RLONG + pos] = score_t;
        if (not_full) fcnt[b] = fc + 1;
      }
    }
    __syncthreads();
    if (doS) {
      const int pos = posS;
      const float* cand = hs + (size_t)(t - KSHORT) * BH + (size_t)b * HID_D;
      const float* ph   = PH + (size_t)(t & 31) * BH + (size_t)b * HID_D;
      float* lm = long_mem + (size_t)pos * BH + (size_t)b * HID_D;
      float* pl = PL       + (size_t)pos * BH + (size_t)b * HID_D;
      for (int i = tid; i < HID_D; i += 256) { lm[i] = cand[i]; pl[i] = ph[i]; }
    }
  }
}

// ---------------------------------------------------------------------------
// Setup kernels
// ---------------------------------------------------------------------------
__global__ void conv_x_kernel(const float* __restrict__ x,
                              half_t* __restrict__ x16, int n) {
  const int i = blockIdx.x * blockDim.x + threadIdx.x;
  if (i < n) x16[i] = (half_t)x[i];
}

// Build TRANSPOSED f16 weights: WcT[n][k] over k in [0,1024) = [Uw;Vw],
// VawT[n][k], UawT[n][k].  One-time cost; makes every WMMA B-fragment a pair
// of contiguous global b128 loads (no LDS staging in the GEMMs).
__global__ void conv_w_kernel(const float* __restrict__ Uw,
                              const float* __restrict__ Vw,
                              const float* __restrict__ Vaw,
                              const float* __restrict__ Uaw,
                              half_t* __restrict__ WcT,
                              half_t* __restrict__ VawT,
                              half_t* __restrict__ UawT) {
  const int i = blockIdx.x * blockDim.x + threadIdx.x;
  if (i < HID_D * (INP_D + HID_D)) {
    const int k = i >> 9;            // 0..1023
    const int n = i & (HID_D - 1);   // 0..511
    const float s = (k < INP_D) ? Uw[(size_t)k * HID_D + n]
                                : Vw[(size_t)(k - INP_D) * HID_D + n];
    WcT[(size_t)n * (INP_D + HID_D) + k] = (half_t)s;
  }
  if (i < HID_D * HID_D) {
    const int k = i >> 9;
    const int n = i & (HID_D - 1);
    VawT[(size_t)n * HID_D + k] = (half_t)Vaw[(size_t)k * HID_D + n];
    UawT[(size_t)n * HID_D + k] = (half_t)Uaw[(size_t)k * HID_D + n];
  }
}

__global__ void init_kernel(float* ls, float* mask, float* buck, int* fc) {
  const int i = blockIdx.x * blockDim.x + threadIdx.x;
  if (i < T_LEN * B_SZ) ls[i] = 0.f;
  if (i < RLONG * B_SZ) mask[i] = NEGM;
  if (i < B_SZ * RLONG) buck[i] = 0.f;
  if (i < B_SZ) fc[i] = 0;
}

// ---------------------------------------------------------------------------
extern "C" void kernel_launch(void* const* d_in, const int* in_sizes, int n_in,
                              void* d_out, int out_size, void* d_ws, size_t ws_size,
                              hipStream_t stream) {
  (void)in_sizes; (void)n_in; (void)out_size;
  const float* x   = (const float*)d_in[0];
  const float* Uw  = (const float*)d_in[1];
  const float* Ub  = (const float*)d_in[2];
  const float* Vw  = (const float*)d_in[3];
  const float* Uaw = (const float*)d_in[4];
  const float* Vaw = (const float*)d_in[5];
  const float* v   = (const float*)d_in[6];
  float* hs = (float*)d_out;

  // ---- carve workspace ----
  char* cur = (char*)d_ws;
  auto alloc = [&](size_t bytes) -> void* {
    void* p = (void*)cur;
    cur += (bytes + 255) & ~(size_t)255;
    return p;
  };
  const size_t BH = (size_t)B_SZ * HID_D;
  half_t* x16   = (half_t*)alloc((size_t)T_LEN * B_SZ * INP_D * 2);
  half_t* WcT   = (half_t*)alloc((size_t)(INP_D + HID_D) * HID_D * 2);
  half_t* VawT  = (half_t*)alloc((size_t)HID_D * HID_D * 2);
  half_t* UawT  = (half_t*)alloc((size_t)HID_D * HID_D * 2);
  half_t* st16  = (half_t*)alloc(BH * 2);
  half_t* h16   = (half_t*)alloc(BH * 2);
  float*  va    = (float*)alloc(BH * 4);
  float*  PH    = (float*)alloc((size_t)KSHORT * BH * 4);
  float*  PL    = (float*)alloc((size_t)RLONG * BH * 4);
  float*  lmem  = (float*)alloc((size_t)RLONG * BH * 4);
  float*  lmask = (float*)alloc((size_t)RLONG * B_SZ * 4);
  float*  lsc   = (float*)alloc((size_t)T_LEN * B_SZ * 4);
  float*  buck  = (float*)alloc((size_t)B_SZ * RLONG * 4);
  int*    fcnt  = (int*)alloc((size_t)B_SZ * 4);
  if ((size_t)(cur - (char*)d_ws) > ws_size) return;  // ws too small

  // ---- hoisted precompute ----
  {
    const int n = T_LEN * B_SZ * INP_D;
    conv_x_kernel<<<(n + 255) / 256, 256, 0, stream>>>(x, x16, n);
    const int nw = HID_D * (INP_D + HID_D);
    conv_w_kernel<<<(nw + 255) / 256, 256, 0, stream>>>(
        Uw, Vw, Vaw, Uaw, WcT, VawT, UawT);
    init_kernel<<<(T_LEN * B_SZ + 255) / 256, 256, 0, stream>>>(
        lsc, lmask, buck, fcnt);
  }

  const int GB = (B_SZ / 16) * (HID_D / 128);  // 32 blocks per GEMM

  // ---- t = 0: fused K=1024 GEMM with zeroed A1 (st_(-1)=0), so
  //      x0@Uw + 0@Vw + Ub == pre;  epilogue: st=pre (raw f16), h=tanh(pre).
  {
    conv_x_kernel<<<((int)BH + 255) / 256, 256, 0, stream>>>(
        (const float*)lsc /*freshly zeroed*/, st16, (int)BH);
    GemmP p0{}; p0.A0 = x16; p0.A1 = st16; p0.WT = WcT; p0.N = HID_D;
    p0.bias = Ub; p0.o32t = hs; p0.o16t = h16; p0.o16r = st16;
    gemm_kernel<INP_D + HID_D, INP_D><<<GB, 128, 0, stream>>>(p0);
    // fused: PH[0] = h0@Uaw  ||  va(t=1) = st0@Vaw
    GemmP pp{}; pp.A0 = h16;  pp.WT = UawT; pp.N = HID_D; pp.o32r = PH;
    GemmP pv{}; pv.A0 = st16; pv.WT = VawT; pv.N = HID_D; pv.o32r = va;
    dual_gemm_kernel<HID_D, HID_D, HID_D, HID_D>
        <<<2 * GB, 128, 0, stream>>>(pp, GB, pv);
  }

  // ---- recurrence: 3 launches per step ----
  for (int t = 1; t < T_LEN; ++t) {
    attn_kernel<<<B_SZ, 256, 0, stream>>>(va, v, hs, PH, PL, lmem, lmask,
                                          lsc, buck, fcnt, st16, t);
    // h_t = tanh([x_t | st_new] @ [Uw;Vw] + Ub)  (K = 1024, fused)
    GemmP ph{}; ph.A0 = x16 + (size_t)t * B_SZ * INP_D; ph.A1 = st16;
    ph.WT = WcT; ph.N = HID_D; ph.bias = Ub;
    ph.o32t = hs + (size_t)t * BH; ph.o16t = h16;
    if (t < T_LEN - 1) {
      GemmP pv{}; pv.A0 = st16; pv.WT = VawT; pv.N = HID_D; pv.o32r = va;
      dual_gemm_kernel<INP_D + HID_D, INP_D, HID_D, HID_D>
          <<<2 * GB, 128, 0, stream>>>(ph, GB, pv);
    } else {
      gemm_kernel<INP_D + HID_D, INP_D><<<GB, 128, 0, stream>>>(ph);
    }
    // PH[t%32] = h_t @ Uaw (projection cache)
    GemmP pr{}; pr.A0 = h16; pr.WT = UawT; pr.N = HID_D;
    pr.o32r = PH + (size_t)(t & 31) * BH;
    gemm_kernel<HID_D, HID_D><<<GB, 128, 0, stream>>>(pr);
  }
}
